// GATLayer_76690936037556
// MI455X (gfx1250) — matrix operand order repaired and checked by reading, not compiled, not run on hardware
//
#include <hip/hip_runtime.h>
#include <hip/hip_fp16.h>

typedef __attribute__((ext_vector_type(16))) _Float16 v16h;
typedef __attribute__((ext_vector_type(8)))  float    v8f;

#define IN_DIM   128
#define OUT_DIM  128
#define HEADS    4
#define HEAD_DIM 32

// order-preserving float <-> uint mapping for atomicMax-based segment max
__device__ __forceinline__ unsigned f2ord(float f) {
    unsigned u = __float_as_uint(f);
    return (u & 0x80000000u) ? ~u : (u | 0x80000000u);
}
__device__ __forceinline__ float ord2f(unsigned u) {
    u = (u & 0x80000000u) ? (u & 0x7fffffffu) : ~u;
    return __uint_as_float(u);
}

// ---------------------------------------------------------------------------
// Kernel 1: h = x @ W^T via v_wmma_f32_16x16x32_f16.
// Block = 256 threads = 8 waves; block b owns output rows [16b, 16b+16),
// wave w owns output cols [16w, 16w+16). K=128 -> 4 chained WMMAs.
// 50000 % 16 == 0, so EXEC is all-ones through the WMMA region.
// A/B lanes use contiguous k = 16*hi + s, same order for both (f16 16x32 / 32x16);
// B[k,n] = W[n,k], so both operands load contiguously from row-major f32.
// ---------------------------------------------------------------------------
__global__ __launch_bounds__(256) void gat_gemm_wmma_f16(
    const float* __restrict__ x, const float* __restrict__ W,
    float* __restrict__ h, int n_nodes)
{
    const int lane = threadIdx.x & 31;
    const int wave = threadIdx.x >> 5;          // 0..7 -> N tile
    const int m0   = blockIdx.x * 16;
    const int n0   = wave * 16;
    const int row  = lane & 15;
    const int hi   = lane >> 4;                 // half-wave selector

    const int mload = min(m0 + row, n_nodes - 1);
    const float* __restrict__ xp = x + (size_t)mload * IN_DIM;
    const float* __restrict__ wp = W + (size_t)(n0 + row) * IN_DIM;

    v8f c = {};
#pragma unroll
    for (int kt = 0; kt < IN_DIM; kt += 32) {
        v16h a, b;
        const int ka = kt + hi * 16;            // A: x[m, ka..ka+15]
#pragma unroll
        for (int j = 0; j < 8; ++j) {
            a[j]     = (_Float16)xp[ka + j];
            a[8 + j] = (_Float16)xp[ka + 8 + j];
        }
        const int kb = kt + hi * 16;            // B: W[n, kb..kb+15]
#pragma unroll
        for (int j = 0; j < 16; ++j) b[j] = (_Float16)wp[kb + j];
        c = __builtin_amdgcn_wmma_f32_16x16x32_f16(
                /*neg_a=*/false, a, /*neg_b=*/false, b,
                /*c_mod=*/(short)0, c, /*reuse_a=*/false, /*reuse_b=*/false);
    }

    // C/D layout: lane (row,hi), VGPR r -> element (M = r + 8*hi, N = row)
#pragma unroll
    for (int r = 0; r < 8; ++r) {
        const int m = m0 + r + hi * 8;
        if (m < n_nodes) h[(size_t)m * OUT_DIM + (n0 + row)] = c[r];
    }
}

// ---------------------------------------------------------------------------
// Kernel 2: per-(node,head) attention scores + init of emax/esum.
// ---------------------------------------------------------------------------
__global__ __launch_bounds__(256) void gat_scores(
    const float* __restrict__ h,
    const float* __restrict__ a_src, const float* __restrict__ a_dst,
    float* __restrict__ s_src, float* __restrict__ s_dst,
    unsigned* __restrict__ emax, float* __restrict__ esum, int n_nodes)
{
    const int idx = blockIdx.x * blockDim.x + threadIdx.x;   // node*HEADS + head
    if (idx >= n_nodes * HEADS) return;
    const int head = idx & (HEADS - 1);
    const float* __restrict__ hp = h + (size_t)(idx >> 2) * OUT_DIM + head * HEAD_DIM;
    const float* __restrict__ as = a_src + head * HEAD_DIM;
    const float* __restrict__ ad = a_dst + head * HEAD_DIM;
    float ss = 0.f, sd = 0.f;
#pragma unroll
    for (int d = 0; d < HEAD_DIM; ++d) {
        const float v = hp[d];
        ss = fmaf(v, as[d], ss);
        sd = fmaf(v, ad[d], sd);
    }
    s_src[idx] = ss;
    s_dst[idx] = sd;
    emax[idx]  = f2ord(-1e9f);     // matches reference's empty-segment fill
    esum[idx]  = 0.f;
}

// ---------------------------------------------------------------------------
// Kernel 3: zero the output accumulator.
// ---------------------------------------------------------------------------
__global__ __launch_bounds__(256) void gat_zero(float* __restrict__ p, int n)
{
    const int i = blockIdx.x * blockDim.x + threadIdx.x;
    if (i < n) p[i] = 0.f;
}

// ---------------------------------------------------------------------------
// Kernel 4: per-(edge,head) logits + leaky-relu + atomic segment max.
// ---------------------------------------------------------------------------
__global__ __launch_bounds__(256) void gat_edge_logits(
    const long long* __restrict__ ei,
    const float* __restrict__ s_src, const float* __restrict__ s_dst,
    float* __restrict__ elog, unsigned* __restrict__ emax, int n_edges)
{
    const int idx = blockIdx.x * blockDim.x + threadIdx.x;   // edge*HEADS + head
    if (idx >= n_edges * HEADS) return;
    const int e    = idx >> 2;
    const int head = idx & (HEADS - 1);
    const int s = (int)ei[e];
    const int d = (int)ei[n_edges + e];
    float v = s_src[s * HEADS + head] + s_dst[d * HEADS + head];
    v = (v > 0.f) ? v : 0.2f * v;                            // leaky_relu(0.2)
    elog[idx] = v;
    atomicMax(&emax[d * HEADS + head], f2ord(v));
}

// ---------------------------------------------------------------------------
// Kernel 5 (fused softmax-numerator + aggregation): per (edge,dim) thread.
//   w = exp(e - emax[dst]) * edge_weight
//   esum[dst,h]  += w                (one lane per head)
//   out[dst,dim] += w * h[src,dim]   (unnormalized accumulate)
// alpha is never materialized; normalization deferred to kernel 6,
// saving an entire 800K-edge pass.
// ---------------------------------------------------------------------------
__global__ __launch_bounds__(256) void gat_edge_agg(
    const long long* __restrict__ ei, const float* __restrict__ ew,
    const float* __restrict__ hbuf, const float* __restrict__ elog,
    const unsigned* __restrict__ emax,
    float* __restrict__ esum, float* __restrict__ out, int n_edges)
{
    const long long idx = (long long)blockIdx.x * blockDim.x + threadIdx.x;
    if (idx >= (long long)n_edges * OUT_DIM) return;
    const int e    = (int)(idx >> 7);
    const int dim  = (int)(idx & (OUT_DIM - 1));
    const int head = dim >> 5;
    const int s = (int)ei[e];
    const int d = (int)ei[n_edges + e];
    const float m = ord2f(emax[d * HEADS + head]);
    const float w = expf(elog[e * HEADS + head] - m) * ew[e];
    if ((dim & (HEAD_DIM - 1)) == 0)
        atomicAdd(&esum[d * HEADS + head], w);
    atomicAdd(&out[(size_t)d * OUT_DIM + dim], w * hbuf[(size_t)s * OUT_DIM + dim]);
}

// ---------------------------------------------------------------------------
// Kernel 6: normalize out by (esum + 1e-12).
// ---------------------------------------------------------------------------
__global__ __launch_bounds__(256) void gat_normalize(
    float* __restrict__ out, const float* __restrict__ esum, int n_nodes)
{
    const int idx = blockIdx.x * blockDim.x + threadIdx.x;
    if (idx >= n_nodes * OUT_DIM) return;
    const int n    = idx >> 7;
    const int head = (idx & (OUT_DIM - 1)) >> 5;
    out[idx] = out[idx] / (esum[n * HEADS + head] + 1e-12f);
}

// ---------------------------------------------------------------------------
// Host launcher. Inputs (setup_inputs order):
// 0=x [N,128] f32, 1=edge_index [2,E] i64, 2=num_nodes scalar,
// 3=edge_weight [E] f32, 4=W [128,128] f32, 5=a_src [4,32] f32, 6=a_dst [4,32] f32
// ---------------------------------------------------------------------------
extern "C" void kernel_launch(void* const* d_in, const int* in_sizes, int n_in,
                              void* d_out, int out_size, void* d_ws, size_t ws_size,
                              hipStream_t stream) {
    const float*     x     = (const float*)d_in[0];
    const long long* ei    = (const long long*)d_in[1];
    const float*     ew    = (const float*)d_in[3];
    const float*     W     = (const float*)d_in[4];
    const float*     a_src = (const float*)d_in[5];
    const float*     a_dst = (const float*)d_in[6];
    float*           out   = (float*)d_out;

    const int n_nodes = in_sizes[0] / IN_DIM;
    const int n_edges = in_sizes[3];

    // workspace layout (floats): h | s_src | s_dst | emax(u32) | esum | elog
    float*    ws    = (float*)d_ws;
    float*    hbuf  = ws;
    float*    s_src = hbuf  + (size_t)n_nodes * OUT_DIM;
    float*    s_dst = s_src + (size_t)n_nodes * HEADS;
    unsigned* emax  = (unsigned*)(s_dst + (size_t)n_nodes * HEADS);
    float*    esum  = (float*)(emax + (size_t)n_nodes * HEADS);
    float*    elog  = esum + (size_t)n_nodes * HEADS;

    const int nblk_gemm = (n_nodes + 15) / 16;
    const int nblk_sc   = (n_nodes * HEADS + 255) / 256;
    const int nblk_zero = (out_size + 255) / 256;
    const int nblk_elog = (n_edges * HEADS + 255) / 256;
    const long long agg_threads = (long long)n_edges * OUT_DIM;
    const int nblk_agg  = (int)((agg_threads + 255) / 256);
    const int nblk_norm = (n_nodes * OUT_DIM + 255) / 256;

    gat_gemm_wmma_f16<<<nblk_gemm, 256, 0, stream>>>(x, W, hbuf, n_nodes);
    gat_scores       <<<nblk_sc,   256, 0, stream>>>(hbuf, a_src, a_dst,
                                                     s_src, s_dst, emax, esum, n_nodes);
    gat_zero         <<<nblk_zero, 256, 0, stream>>>(out, out_size);
    gat_edge_logits  <<<nblk_elog, 256, 0, stream>>>(ei, s_src, s_dst, elog, emax, n_edges);
    gat_edge_agg     <<<nblk_agg,  256, 0, stream>>>(ei, ew, hbuf, elog, emax,
                                                     esum, out, n_edges);
    gat_normalize    <<<nblk_norm, 256, 0, stream>>>(out, esum, n_nodes);
}
